// MHA_17927193494086
// MI455X (gfx1250) — matrix operand order repaired and checked by reading, not compiled, not run on hardware
//
#include <hip/hip_runtime.h>
#include <hip/hip_bf16.h>
#include <math.h>

#define BS 2
#define NN 512
#define DM 128
#define NH 8
#define DH 16

typedef __attribute__((ext_vector_type(16))) __bf16 v16bf;
typedef __attribute__((ext_vector_type(2)))  __bf16 v2bf;
typedef __attribute__((ext_vector_type(8)))  float  v8f;

union Frag16 { unsigned int u[8]; v16bf v; };

// native f32->bf16 convert (hardware cvt on gfx1250), packed pair -> u32
__device__ __forceinline__ unsigned int pkbf(float x, float y) {
  v2bf v;
  v.x = (__bf16)x;
  v.y = (__bf16)y;
  return __builtin_bit_cast(unsigned int, v);
}
__device__ __forceinline__ unsigned short bf_bits(float f) {
  __bf16 b = (__bf16)f;
  return __builtin_bit_cast(unsigned short, b);
}

// wave32 XOR-butterfly reductions via ds_swizzle (group-of-32: and=0x1f, or=0, xor=MASK)
template<int MASK>
__device__ __forceinline__ float xor_add(float x) {
  int o = __builtin_amdgcn_ds_swizzle(__float_as_int(x), (MASK << 10) | 0x1f);
  return x + __int_as_float(o);
}
template<int MASK>
__device__ __forceinline__ float xor_max(float x) {
  int o = __builtin_amdgcn_ds_swizzle(__float_as_int(x), (MASK << 10) | 0x1f);
  return fmaxf(x, __int_as_float(o));
}

// ---------------------------------------------------------------------------
// K1: node projections. out = x @ W.T for 5 weights.
// Q,K,NI,NJ kept f32 row-major; V stored bf16 transposed [b][c][j] for K3.
// One wave per 16x16 output tile, 4 x v_wmma_f32_16x16x32_bf16.
// ---------------------------------------------------------------------------
__global__ __launch_bounds__(32)
void proj_kernel(const float* __restrict__ hIn,
                 const float* __restrict__ Wq, const float* __restrict__ Wk,
                 const float* __restrict__ Wv, const float* __restrict__ Wni,
                 const float* __restrict__ Wnj,
                 float* __restrict__ Qf, float* __restrict__ Kf,
                 unsigned short* __restrict__ Vt,
                 float* __restrict__ NI, float* __restrict__ NJ)
{
  const int widx = blockIdx.z;
  const float* W = (widx == 0) ? Wq : (widx == 1) ? Wk : (widx == 2) ? Wv
                 : (widx == 3) ? Wni : Wnj;
  const int m0 = blockIdx.x * 16;          // row tile in flattened (bs*n)
  const int n0 = blockIdx.y * 16;          // output-column tile
  const int lane = threadIdx.x & 31;
  const int t  = lane & 15;
  const int hw = lane >> 4;

  // B fragments: B[k][n] = W[n0+t][k]  (lane holds K = ks*32 + hw*16 + e)
  Frag16 bf[4];
  const float* wrow = W + (size_t)(n0 + t) * DM;
#pragma unroll
  for (int ks = 0; ks < 4; ++ks) {
    const int kb = ks * 32 + hw * 16;
#pragma unroll
    for (int e2 = 0; e2 < 8; ++e2) {
      float2 f = *(const float2*)(wrow + kb + 2 * e2);
      bf[ks].u[e2] = pkbf(f.x, f.y);
    }
  }

  // A fragments: A[m][k] = h[m0+m][k], m = lane&15
  const float* hrow = hIn + (size_t)(m0 + t) * DM;
  v8f c = {};
#pragma unroll
  for (int ks = 0; ks < 4; ++ks) {
    Frag16 af;
#pragma unroll
    for (int p = 0; p < 8; ++p) {
      const int kk = ks * 32 + hw * 16 + 2 * p;
      float2 f = *(const float2*)(hrow + kk);
      af.u[p] = pkbf(f.x, f.y);
    }
    c = __builtin_amdgcn_wmma_f32_16x16x32_bf16(false, af.v, false, bf[ks].v,
                                                (short)0, c, false, false);
  }

#pragma unroll
  for (int r = 0; r < 8; ++r) {
    const int m  = m0 + r + hw * 8;        // C: M = r + 8*half
    const int cc = n0 + t;                 // C: N = lane&15
    const float val = c[r];
    if (widx == 0)      Qf[(size_t)m * DM + cc] = val;
    else if (widx == 1) Kf[(size_t)m * DM + cc] = val;
    else if (widx == 2) {
      const int bidx = m >> 9, nrow = m & 511;
      Vt[((size_t)(bidx * DM + cc)) * NN + nrow] = bf_bits(val);
    }
    else if (widx == 3) NI[(size_t)m * DM + cc] = val;
    else                NJ[(size_t)m * DM + cc] = val;
  }
}

// ---------------------------------------------------------------------------
// K2: edge projection + broadcasts + edge_out store + fused scores.
// Block = (b, 16 i's, 16 j's); e tile staged once in LDS (bf16, 64 KB),
// 8 waves = 8 heads reuse it. Each wave: 16 i-rows x 4 WMMA k-steps.
// ---------------------------------------------------------------------------
__global__ __launch_bounds__(256)
void edge_kernel(const float* __restrict__ eIn, const float* __restrict__ We,
                 const float* __restrict__ Qf, const float* __restrict__ Kf,
                 const float* __restrict__ NI, const float* __restrict__ NJ,
                 float* __restrict__ edge_out, float* __restrict__ scores)
{
  __shared__ unsigned int lds_e[256 * 64];  // 256 (ii,jj) rows x 128 bf16 = 64 KB
  const int j0 = blockIdx.x * 16;
  const int i0 = blockIdx.y * 16;
  const int b  = blockIdx.z;
  const int tid  = threadIdx.x;
  const int lane = tid & 31;
  const int hd   = tid >> 5;               // wave index == head
  const int t  = lane & 15;
  const int hw = lane >> 4;

  // cooperative stage: 16x16x128 f32 -> packed bf16 in LDS
#pragma unroll 4
  for (int it = 0; it < 32; ++it) {
    const int q   = tid + it * 256;        // 0..8191 float4 slots
    const int row = q >> 5;                // (ii*16+jj)
    const int c4  = (q & 31) * 4;
    const int ii = row >> 4, jj = row & 15;
    const float4 f = *(const float4*)(eIn +
        (((size_t)(b * NN + i0 + ii) * NN + (j0 + jj)) * DM + c4));
    lds_e[row * 64 + (c4 >> 1)]     = pkbf(f.x, f.y);
    lds_e[row * 64 + (c4 >> 1) + 1] = pkbf(f.z, f.w);
  }
  __syncthreads();

  // per-head B fragments from We rows [hd*16 .. hd*16+15] (L2-resident)
  Frag16 bf[4];
  const float* wrow = We + (size_t)(hd * DH + t) * DM;
#pragma unroll
  for (int ks = 0; ks < 4; ++ks) {
    const int kb = ks * 32 + hw * 16;
#pragma unroll
    for (int e2 = 0; e2 < 8; ++e2) {
      float2 f = *(const float2*)(wrow + kb + 2 * e2);
      bf[ks].u[e2] = pkbf(f.x, f.y);
    }
  }

  for (int ii = 0; ii < 16; ++ii) {
    const int i = i0 + ii;
    v8f c = {};
#pragma unroll
    for (int ks = 0; ks < 4; ++ks) {
      Frag16 af;
      const int arow = ii * 16 + t;        // A: M(=local jj) = lane&15
#pragma unroll
      for (int p = 0; p < 8; ++p) {
        const int w = ks * 16 + hw * 8 + p;
        af.u[p] = lds_e[arow * 64 + w];
      }
      c = __builtin_amdgcn_wmma_f32_16x16x32_bf16(false, af.v, false, bf[ks].v,
                                                  (short)0, c, false, false);
    }

    const size_t ni_idx = ((size_t)(b * NN + i)) * DM + hd * DH + t;
    const float niv = NI[ni_idx];
    const float qv  = Qf[ni_idx];
    float srow[8];
#pragma unroll
    for (int r = 0; r < 8; ++r) {
      const int jj = r + hw * 8;           // C: M = r + 8*half, N = lane&15
      const int j  = j0 + jj;
      const size_t nidx = ((size_t)(b * NN + j)) * DM + hd * DH + t;
      const float val = c[r] + niv + NJ[nidx];
      edge_out[(((size_t)(b * NN + i)) * NN + j) * DM + hd * DH + t] = val;
      srow[r] = val * qv * Kf[nidx];       // q_id * ev_ijd * k_jd
    }
    // reduce over d (16 N-lanes, XOR stays within lane-half so jj is preserved)
#pragma unroll
    for (int r = 0; r < 8; ++r) {
      float s = srow[r];
      s = xor_add<1>(s); s = xor_add<2>(s); s = xor_add<4>(s); s = xor_add<8>(s);
      srow[r] = s * 0.25f;                 // 1/sqrt(dh)
    }
    if (t == 0) {
#pragma unroll
      for (int r = 0; r < 8; ++r) {
        const int j = j0 + r + hw * 8;
        scores[(((size_t)(b * NH + hd)) * NN + i) * NN + j] = srow[r];
      }
    }
  }
}

// ---------------------------------------------------------------------------
// K3: softmax over j (512) + attn @ V via 16 WMMA k-steps per (b,h,i-tile).
// ---------------------------------------------------------------------------
__global__ __launch_bounds__(256)
void softmax_av_kernel(const float* __restrict__ scores,
                       const unsigned short* __restrict__ Vt,
                       float* __restrict__ outp)
{
  __shared__ unsigned int lds_p[16 * 256]; // probs, packed bf16 pairs [row][jword]
  const int i0 = blockIdx.x * 16;
  const int hd = blockIdx.y;
  const int b  = blockIdx.z;
  const int tid  = threadIdx.x;
  const int lane = tid & 31;
  const int wave = tid >> 5;

  // each wave softmaxes rows {wave, wave+8}; 16 scores per lane
#pragma unroll
  for (int rr = 0; rr < 2; ++rr) {
    const int row = wave + rr * 8;
    const int i = i0 + row;
    const float* sr = scores + (((size_t)(b * NH + hd) * NN + i) * NN) + lane * 16;
    float vals[16];
#pragma unroll
    for (int q4 = 0; q4 < 4; ++q4) {
      float4 f = *(const float4*)(sr + q4 * 4);
      vals[q4*4+0]=f.x; vals[q4*4+1]=f.y; vals[q4*4+2]=f.z; vals[q4*4+3]=f.w;
    }
    float mx = vals[0];
#pragma unroll
    for (int q = 1; q < 16; ++q) mx = fmaxf(mx, vals[q]);
    mx = xor_max<1>(mx); mx = xor_max<2>(mx); mx = xor_max<4>(mx);
    mx = xor_max<8>(mx); mx = xor_max<16>(mx);
    float sum = 0.f;
#pragma unroll
    for (int q = 0; q < 16; ++q) { vals[q] = __expf(vals[q] - mx); sum += vals[q]; }
    sum = xor_add<1>(sum); sum = xor_add<2>(sum); sum = xor_add<4>(sum);
    sum = xor_add<8>(sum); sum = xor_add<16>(sum);
    const float inv = 1.0f / sum;
#pragma unroll
    for (int p = 0; p < 8; ++p)
      lds_p[row * 256 + lane * 8 + p] = pkbf(vals[2*p] * inv, vals[2*p+1] * inv);
  }
  __syncthreads();

  if (wave == 0) {                          // out tile = attn(16x512) @ V(512x16)
    const int t  = lane & 15;
    const int hw = lane >> 4;
    v8f c = {};
    for (int ks = 0; ks < 16; ++ks) {
      Frag16 af, bfv;
#pragma unroll
      for (int p = 0; p < 8; ++p) {
        const int w = ks * 16 + hw * 8 + p;
        af.u[p] = lds_p[t * 256 + w];       // A row m = lane&15
      }
      const unsigned int* vb = (const unsigned int*)
        (Vt + ((size_t)(b * DM + hd * DH + t)) * NN + ks * 32 + hw * 16);
#pragma unroll
      for (int p = 0; p < 8; ++p) bfv.u[p] = vb[p];   // contiguous j (transposed V)
      c = __builtin_amdgcn_wmma_f32_16x16x32_bf16(false, af.v, false, bfv.v,
                                                  (short)0, c, false, false);
    }
#pragma unroll
    for (int r = 0; r < 8; ++r) {
      const int i = i0 + r + hw * 8;
      outp[((size_t)(b * NN + i)) * DM + hd * DH + t] = c[r];
    }
  }
}

extern "C" void kernel_launch(void* const* d_in, const int* in_sizes, int n_in,
                              void* d_out, int out_size, void* d_ws, size_t ws_size,
                              hipStream_t stream) {
  const float* hIn = (const float*)d_in[0];
  const float* eIn = (const float*)d_in[1];
  const float* Wq  = (const float*)d_in[2];
  const float* Wk  = (const float*)d_in[3];
  const float* Wv  = (const float*)d_in[4];
  const float* We  = (const float*)d_in[5];
  const float* Wni = (const float*)d_in[6];
  const float* Wnj = (const float*)d_in[7];

  float* outp     = (float*)d_out;                 // (2,512,128)
  float* edge_out = outp + (size_t)BS * NN * DM;   // (2,512,512,128)

  const size_t nproj = (size_t)BS * NN * DM;       // 131072
  float* Qf = (float*)d_ws;
  float* Kf = Qf + nproj;
  float* NI = Kf + nproj;
  float* NJ = NI + nproj;
  unsigned short* Vt = (unsigned short*)(NJ + nproj);     // bf16 transposed V
  float* scores = (float*)(Vt + nproj);                   // bs*H*n*n f32 (~16.8 MB)

  proj_kernel<<<dim3(64, 8, 5), dim3(32), 0, stream>>>(
      hIn, Wq, Wk, Wv, Wni, Wnj, Qf, Kf, Vt, NI, NJ);
  edge_kernel<<<dim3(32, 32, 2), dim3(256), 0, stream>>>(
      eIn, We, Qf, Kf, NI, NJ, edge_out, scores);
  softmax_av_kernel<<<dim3(32, 8, 2), dim3(256), 0, stream>>>(
      scores, Vt, outp);
}
